// MaskGRU_27865747816714
// MI455X (gfx1250) — compile-verified, matrix operand order checked by reading, not run on hardware
//
#include <hip/hip_runtime.h>
#include <hip/hip_bf16.h>

// Problem dims (fixed by the reference)
#define B_  64
#define T_  512
#define I_  256
#define H_  256
#define G3  768      // 3*H
#define NWG 16       // persistent workgroups in recurrent phase

typedef __attribute__((ext_vector_type(16))) _Float16 v16h;
typedef __attribute__((ext_vector_type(8)))  _Float16 v8h;
typedef __attribute__((ext_vector_type(8)))  float    v8f;
typedef __attribute__((ext_vector_type(4)))  float    v4f;

static __device__ __forceinline__ v16h combine8(v8h lo, v8h hi) {
    v16h r;
#pragma unroll
    for (int e = 0; e < 8; ++e) { r[e] = lo[e]; r[8 + e] = hi[e]; }
    return r;
}

// Load 16 f32 (two contiguous 8-element runs at k0a / k0b) and convert to f16,
// matching the 16-bit A/B WMMA per-lane layout.
static __device__ __forceinline__ v16h load_a16_from_f32(const float* __restrict__ row,
                                                         int k0a, int k0b) {
    v4f a0 = *(const v4f*)(row + k0a);
    v4f a1 = *(const v4f*)(row + k0a + 4);
    v4f a2 = *(const v4f*)(row + k0b);
    v4f a3 = *(const v4f*)(row + k0b + 4);
    v16h r;
#pragma unroll
    for (int e = 0; e < 4; ++e) {
        r[e]      = (_Float16)a0[e];
        r[4 + e]  = (_Float16)a1[e];
        r[8 + e]  = (_Float16)a2[e];
        r[12 + e] = (_Float16)a3[e];
    }
    return r;
}

// ---------------------------------------------------------------------------
// Phase 0: zero the h ping-pong buffer and the grid-sync counter (per launch,
// graph-replay safe: runs every call on `stream`).
// ---------------------------------------------------------------------------
__global__ void init_kernel(unsigned* __restrict__ hbuf_u32, unsigned* __restrict__ sync) {
    const int nu = (2 * B_ * H_ * 2) / 4;  // two f16 [B,H] buffers in dwords
    for (int i = blockIdx.x * blockDim.x + threadIdx.x; i < nu; i += blockDim.x * gridDim.x)
        hbuf_u32[i] = 0u;
    if (blockIdx.x == 0 && threadIdx.x == 0) *sync = 0u;
}

// ---------------------------------------------------------------------------
// Phase 1: xp[b*T+t][g] = sum_k x[b,t,k] * W_ih[g,k] + b_ih[g]
// M=32768, N=768, K=256. One 16x16 tile per wave, 8 WMMA (K chunks) each.
// ---------------------------------------------------------------------------
__global__ __launch_bounds__(256) void xproj_kernel(const float* __restrict__ x,
                                                    const float* __restrict__ W_ih,
                                                    const float* __restrict__ b_ih,
                                                    float* __restrict__ xp) {
    const int lane  = threadIdx.x & 31;
    const int wave  = threadIdx.x >> 5;
    const int half  = lane >> 4;
    const int l15   = lane & 15;
    const int mtile = blockIdx.x * 8 + wave;   // 0..2047
    const int ntile = blockIdx.y;              // 0..47

    const float* arow = x    + (size_t)(mtile * 16 + l15) * I_;
    const float* brow = W_ih + (size_t)(ntile * 16 + l15) * I_;

    v8f acc = {};
#pragma unroll
    for (int kc = 0; kc < 8; ++kc) {
        const int k0a = kc * 32 + half * 8;
        const int k0b = k0a + 16;
        v16h a = load_a16_from_f32(arow, k0a, k0b);
        v16h b = load_a16_from_f32(brow, k0a, k0b);
        acc = __builtin_amdgcn_wmma_f32_16x16x32_f16(false, a, false, b,
                                                     (short)0, acc, false, false);
    }
    const int   n    = ntile * 16 + l15;
    const float bias = b_ih[n];
#pragma unroll
    for (int r = 0; r < 8; ++r) {
        const int m = mtile * 16 + r + 8 * half;
        xp[(size_t)m * G3 + n] = acc[r] + bias;
    }
}

// ---------------------------------------------------------------------------
// Phase 2: persistent bidirectional recurrence.
// 16 WGs x 192 threads (6 waves). WG j owns h columns [16j, 16j+16) and gate
// columns {16j.., 256+16j.., 512+16j..}. W_hh slice held in registers (v16h
// Breg[8] per wave) for all 1024 steps. h ping-pongs between two f16 global
// buffers; one atomic grid barrier per step.
// ---------------------------------------------------------------------------
__global__ __launch_bounds__(192) void gru_rec_kernel(const float* __restrict__ xp,
                                                      const float* __restrict__ W_hh,
                                                      const float* __restrict__ b_hh,
                                                      const float* __restrict__ mask,
                                                      _Float16* __restrict__ hglob,   // 2 x [B,H] f16
                                                      unsigned* __restrict__ sync,
                                                      float* __restrict__ out,        // [B,T,2H]
                                                      float* __restrict__ hn) {       // [B,2H]
    __shared__ _Float16 wlds[3][16][I_];   // W_hh slice as f16 (24 KB)
    __shared__ float    hp[B_][48];        // gate pre-activations (12 KB)
    __shared__ float    hcur[B_][16];      // owned f32 h slice (4 KB)
    __shared__ float    bhh[48];

    const int j    = blockIdx.x;           // 0..15
    const int tid  = threadIdx.x;
    const int lane = tid & 31;
    const int wave = tid >> 5;             // 0..5
    const int half = lane >> 4;
    const int l15  = lane & 15;
    const int nt   = wave >> 1;            // gate tile 0..2
    const int mp   = (wave & 1) * 2;       // m-tiles {mp, mp+1}

    // Stage W_hh slice -> LDS (f16), biases, zero h
    for (int idx = tid; idx < 3 * 16 * I_; idx += blockDim.x) {
        const int g = idx / (16 * I_);
        const int r = idx - g * (16 * I_);
        const int c = r / I_;
        const int k = r - c * I_;
        wlds[g][c][k] = (_Float16)W_hh[((size_t)g * H_ + j * 16 + c) * H_ + k];
    }
    for (int idx = tid; idx < 48; idx += blockDim.x)
        bhh[idx] = b_hh[(idx >> 4) * H_ + j * 16 + (idx & 15)];
    for (int idx = tid; idx < B_ * 16; idx += blockDim.x)
        hcur[idx >> 4][idx & 15] = 0.0f;
    __syncthreads();

    // B operand: resident in registers for the whole recurrence
    v16h Breg[8];
    {
        const _Float16* wbase = &wlds[nt][l15][0];
#pragma unroll
        for (int kc = 0; kc < 8; ++kc) {
            const int k0a = kc * 32 + half * 8;
            Breg[kc] = combine8(*(const v8h*)(wbase + k0a),
                                *(const v8h*)(wbase + k0a + 16));
        }
    }

    unsigned steps_done = 0;
    for (int dir = 0; dir < 2; ++dir) {
        for (int s = 0; s < T_; ++s) {
            const int t = dir ? (T_ - 1 - s) : s;
            const int p = (int)(steps_done & 1u);
            const _Float16* hrd = hglob + (size_t)p * (B_ * H_);
            _Float16*       hwr = hglob + (size_t)(p ^ 1) * (B_ * H_);

            // ---- hp = h @ W_hh^T (48 owned columns), 16 WMMAs per wave ----
            v8f acc0 = {}, acc1 = {};
            const _Float16* r0 = hrd + (size_t)((mp + 0) * 16 + l15) * H_;
            const _Float16* r1 = hrd + (size_t)((mp + 1) * 16 + l15) * H_;
#pragma unroll
            for (int kc = 0; kc < 8; ++kc) {
                const int k0a = kc * 32 + half * 8;
                const int k0b = k0a + 16;
                v16h a0 = combine8(*(const v8h*)(r0 + k0a), *(const v8h*)(r0 + k0b));
                v16h a1 = combine8(*(const v8h*)(r1 + k0a), *(const v8h*)(r1 + k0b));
                acc0 = __builtin_amdgcn_wmma_f32_16x16x32_f16(false, a0, false, Breg[kc],
                                                              (short)0, acc0, false, false);
                acc1 = __builtin_amdgcn_wmma_f32_16x16x32_f16(false, a1, false, Breg[kc],
                                                              (short)0, acc1, false, false);
            }
#pragma unroll
            for (int r = 0; r < 8; ++r) {
                hp[(mp + 0) * 16 + r + 8 * half][nt * 16 + l15] = acc0[r];
                hp[(mp + 1) * 16 + r + 8 * half][nt * 16 + l15] = acc1[r];
            }
            __syncthreads();

            // ---- fused gates over owned [64 x 16] slice ----
            for (int idx = tid; idx < B_ * 16; idx += blockDim.x) {
                const int b = idx >> 4, c = idx & 15;
                const float hr = hp[b][c]      + bhh[c];
                const float hz = hp[b][16 + c] + bhh[16 + c];
                const float hv = hp[b][32 + c] + bhh[32 + c];
                const float* xrow = xp + ((size_t)b * T_ + t) * G3 + j * 16 + c;
                const float xr = xrow[0], xz = xrow[H_], xn = xrow[2 * H_];
                const float rg = 1.0f / (1.0f + __expf(-(xr + hr)));
                const float zg = 1.0f / (1.0f + __expf(-(xz + hz)));
                const float ng = tanhf(xn + rg * hv);
                const float hold = hcur[b][c];
                const float hnew = (1.0f - zg) * ng + zg * hold;
                const float mt   = mask[(size_t)b * T_ + t];
                const float hb   = mt * hnew + (1.0f - mt) * hold;
                hcur[b][c] = hb;
                hwr[(size_t)b * H_ + j * 16 + c] = (_Float16)hb;
                out[((size_t)b * T_ + t) * (2 * H_) + dir * H_ + j * 16 + c] = hb;
            }

            // ---- grid-wide barrier (16 persistent WGs) ----
            __threadfence();
            __syncthreads();
            ++steps_done;
            if (tid == 0) {
                atomicAdd(sync, 1u);
                const unsigned target = (unsigned)NWG * steps_done;
                while (atomicAdd(sync, 0u) < target) { __builtin_amdgcn_s_sleep(1); }
            }
            __syncthreads();
        }
        // capture h_n for this direction
        for (int idx = tid; idx < B_ * 16; idx += blockDim.x) {
            const int b = idx >> 4, c = idx & 15;
            hn[(size_t)b * (2 * H_) + dir * H_ + j * 16 + c] = hcur[b][c];
        }
    }
}

// ---------------------------------------------------------------------------
extern "C" void kernel_launch(void* const* d_in, const int* in_sizes, int n_in,
                              void* d_out, int out_size, void* d_ws, size_t ws_size,
                              hipStream_t stream) {
    const float* x    = (const float*)d_in[0];
    const float* mask = (const float*)d_in[1];
    const float* W_ih = (const float*)d_in[2];
    const float* W_hh = (const float*)d_in[3];
    const float* b_ih = (const float*)d_in[4];
    const float* b_hh = (const float*)d_in[5];

    const size_t XP_BYTES = (size_t)B_ * T_ * G3 * sizeof(float);     // ~96 MB
    const size_t H_BYTES  = (size_t)2 * B_ * H_ * sizeof(_Float16);   // 64 KB ping-pong

    float*    xp    = (float*)d_ws;
    _Float16* hbuf  = (_Float16*)((char*)d_ws + XP_BYTES);
    unsigned* syncp = (unsigned*)((char*)d_ws + XP_BYTES + H_BYTES);

    float* out = (float*)d_out;
    float* hn  = out + (size_t)B_ * T_ * 2 * H_;

    init_kernel<<<32, 256, 0, stream>>>((unsigned*)hbuf, syncp);
    xproj_kernel<<<dim3((B_ * T_ / 16) / 8, G3 / 16), 256, 0, stream>>>(x, W_ih, b_ih, xp);
    gru_rec_kernel<<<NWG, 192, 0, stream>>>(xp, W_hh, b_hh, mask, hbuf, syncp, out, hn);
}
